// LSTMGraphBlock_64476049047622
// MI455X (gfx1250) — matrix operand order repaired and checked by reading, not compiled, not run on hardware
//
#include <hip/hip_runtime.h>
#include <stdint.h>

// N=50000, D=128, E=800000
#define D    128
#define KTOT 384   // A = [h | x | agg]
#define CTOT 512   // 4 gates x 128 output cols

typedef __attribute__((ext_vector_type(16))) _Float16 v16h;
typedef __attribute__((ext_vector_type(8)))  float    v8f;

union AFrag { v16h h; uint32_t u[8]; };
union BFrag { v16h h; uint4 q[2]; };

struct WeightPtrs {
    const float* Wg[4];
    const float* bg[4];
    const float* Wl[4];
    const float* bl[4];
    const float* Wr[4];
};

// ---------------------------------------------------------------- zero ws
__global__ void zero_ws(float* p, long long n) {
    long long i = (long long)blockIdx.x * blockDim.x + threadIdx.x;
    long long stride = (long long)gridDim.x * blockDim.x;
    for (; i < n; i += stride) p[i] = 0.0f;
}

// ------------------------------------------- fuse + transpose weights (f16)
// WcombT[c][k], c = gate*128 + j, k in [0,384):
//   k <128  : Wg[g][k][j] + Wr[g][k][j]      (h path)
//   k <256  : Wg[g][k][j]                    (x path; Wg rows 128..255)
//   else    : Wl[g][k-256][j]                (agg path)
__global__ void prep_weights(WeightPtrs W, _Float16* __restrict__ WcombT,
                             float* __restrict__ biasComb) {
    int idx = blockIdx.x * blockDim.x + threadIdx.x;
    if (idx < CTOT) {
        int g = idx >> 7, j = idx & 127;
        biasComb[idx] = W.bg[g][j] + W.bl[g][j];
    }
    if (idx >= CTOT * KTOT) return;
    int c = idx / KTOT, k = idx - c * KTOT;
    int g = c >> 7, j = c & 127;
    float v;
    if (k < 128)      v = W.Wg[g][k * D + j] + W.Wr[g][k * D + j];
    else if (k < 256) v = W.Wg[g][k * D + j];
    else              v = W.Wl[g][(k - 256) * D + j];
    WcombT[(size_t)c * KTOT + k] = (_Float16)v;
}

// ------------------------------------------------- edge scatter (mean agg)
// one wave32 per edge: lane l moves cols [4l,4l+4)
__global__ void edge_scatter(const float* __restrict__ h, const int* __restrict__ ei,
                             float* __restrict__ agg, float* __restrict__ deg,
                             int E_, int N_) {
    int wave = blockIdx.x * (blockDim.x >> 5) + (threadIdx.x >> 5);
    int lane = threadIdx.x & 31;
    if (wave >= E_) return;
    int src = ei[wave];
    int dst = ei[E_ + wave];
    const float4 v = *(const float4*)(h + (size_t)src * D + lane * 4);
    float* a = agg + (size_t)dst * D + lane * 4;
    atomicAdd(a + 0, v.x);
    atomicAdd(a + 1, v.y);
    atomicAdd(a + 2, v.z);
    atomicAdd(a + 3, v.w);
    if (lane == 0) atomicAdd(deg + dst, 1.0f);
}

__global__ void agg_div(float* __restrict__ agg, const float* __restrict__ deg, int N_) {
    int i = blockIdx.x * blockDim.x + threadIdx.x;
    if (i >= N_ * D) return;
    agg[i] = agg[i] / fmaxf(deg[i >> 7], 1.0f);
}

// -------------------------------------------- fused GEMM + LSTM + LayerNorm
// block = 128 threads = 4 waves; wave g computes gate g for a 32-row tile.
// Per wave: acc[2 m-subtiles][8 n-tiles], K = 384 = 12 steps of 32 (f16 WMMA).
__global__ __launch_bounds__(128)
void fused_gates(const float* __restrict__ h, const float* __restrict__ x,
                 const float* __restrict__ agg, const float* __restrict__ cellp,
                 const _Float16* __restrict__ WcombT, const float* __restrict__ biasComb,
                 const float* __restrict__ gamma, const float* __restrict__ beta,
                 float* __restrict__ out_h, float* __restrict__ out_c, int N_) {
    __shared__ _Float16 Atile[32 * KTOT];   // 24 KB: [h|x|agg] rows as f16
    __shared__ _Float16 Gtile[4 * 32 * D];  // 32 KB: activated gate values

    const int tid  = threadIdx.x;
    const int lane = tid & 31;
    const int g    = tid >> 5;              // wave id == gate (f,i,c,o)
    const int row0 = blockIdx.x * 32;

    // ---- stage A-panel (fp32 -> f16), float4-vectorized, zero-pad tail rows
    for (int v = tid; v < 32 * (KTOT / 4); v += 128) {
        int m  = v / (KTOT / 4);
        int k4 = (v - m * (KTOT / 4)) * 4;
        int row = row0 + m;
        float4 f = make_float4(0.f, 0.f, 0.f, 0.f);
        if (row < N_) {
            const float* srcp;
            if (k4 < D)          srcp = h   + (size_t)row * D + k4;
            else if (k4 < 2 * D) srcp = x   + (size_t)row * D + (k4 - D);
            else                 srcp = agg + (size_t)row * D + (k4 - 2 * D);
            f = *(const float4*)srcp;
        }
        _Float16* dp = &Atile[m * KTOT + k4];
        dp[0] = (_Float16)f.x; dp[1] = (_Float16)f.y;
        dp[2] = (_Float16)f.z; dp[3] = (_Float16)f.w;
    }
    __syncthreads();

    // ---- WMMA main loop
    v8f acc[2][8];
    const v8f vzero = {0.f, 0.f, 0.f, 0.f, 0.f, 0.f, 0.f, 0.f};
#pragma unroll
    for (int s = 0; s < 2; ++s)
#pragma unroll
        for (int nt = 0; nt < 8; ++nt) acc[s][nt] = vzero;

    const int khalf = lane >> 4;    // A: lanes 16-31 take K+8; B: lanes 16-31 take K+8
    const int mbase = lane & 15;
    const _Float16* wbase =
        WcombT + (size_t)(g * 128 + (lane & 15)) * KTOT + khalf * 8;

    for (int kt = 0; kt < 12; ++kt) {
        AFrag a[2];
#pragma unroll
        for (int s = 0; s < 2; ++s) {
            const _Float16* ap = &Atile[(s * 16 + mbase) * KTOT + kt * 32 + khalf * 8];
#pragma unroll
            for (int p = 0; p < 8; ++p) {
                int koff = ((p >> 2) << 4) + ((p & 3) << 1);   // K pairs per ISA layout
                a[s].u[p] = *(const uint32_t*)(ap + koff);
            }
        }
#pragma unroll
        for (int nt = 0; nt < 8; ++nt) {
            BFrag b;
            const _Float16* bp = wbase + (size_t)nt * 16 * KTOT + kt * 32;
            b.q[0] = *(const uint4*)(bp);
            b.q[1] = *(const uint4*)(bp + 16);
            acc[0][nt] = __builtin_amdgcn_wmma_f32_16x16x32_f16(
                false, a[0].h, false, b.h, (short)0, acc[0][nt], false, false);
            acc[1][nt] = __builtin_amdgcn_wmma_f32_16x16x32_f16(
                false, a[1].h, false, b.h, (short)0, acc[1][nt], false, false);
        }
    }

    // ---- bias + activation, exchange gates via LDS
    const int nlane = lane & 15;
    const int mh    = (lane >> 4) << 3;    // C/D: lanes 16-31 hold M+8
#pragma unroll
    for (int nt = 0; nt < 8; ++nt) {
        int col = nt * 16 + nlane;
        float bias = biasComb[g * 128 + col];
#pragma unroll
        for (int s = 0; s < 2; ++s) {
#pragma unroll
            for (int r = 0; r < 8; ++r) {
                float pre = acc[s][nt][r] + bias;
                float act = (g == 2) ? tanhf(pre)
                                     : 1.0f / (1.0f + __expf(-pre));
                int m = s * 16 + mh + r;
                Gtile[(g * 32 + m) * D + col] = (_Float16)act;
            }
        }
    }
    __syncthreads();

    // ---- LSTM pointwise + LayerNorm: 4 lanes per row, 32 cols per lane
    const int mrow = tid >> 2;
    const int q    = tid & 3;
    const int row  = row0 + mrow;
    if (row < N_) {
        float hv[32];
        float sum = 0.f, sumsq = 0.f;
#pragma unroll
        for (int j = 0; j < 32; ++j) {
            int col = q * 32 + j;
            float f = (float)Gtile[(0 * 32 + mrow) * D + col];
            float i = (float)Gtile[(1 * 32 + mrow) * D + col];
            float c = (float)Gtile[(2 * 32 + mrow) * D + col];
            float o = (float)Gtile[(3 * 32 + mrow) * D + col];
            float cn = f * cellp[(size_t)row * D + col] + i * c;
            out_c[(size_t)row * D + col] = cn;
            float v = o * tanhf(cn);
            hv[j] = v;
            sum += v; sumsq += v * v;
        }
        sum   += __shfl_xor(sum,   1); sum   += __shfl_xor(sum,   2);
        sumsq += __shfl_xor(sumsq, 1); sumsq += __shfl_xor(sumsq, 2);
        float mu   = sum * (1.0f / 128.0f);
        float var  = sumsq * (1.0f / 128.0f) - mu * mu;
        float rstd = rsqrtf(var + 1e-5f);
#pragma unroll
        for (int j = 0; j < 32; ++j) {
            int col = q * 32 + j;
            out_h[(size_t)row * D + col] = (hv[j] - mu) * rstd * gamma[col] + beta[col];
        }
    }
}

// ---------------------------------------------------------------- launcher
extern "C" void kernel_launch(void* const* d_in, const int* in_sizes, int n_in,
                              void* d_out, int out_size, void* d_ws, size_t ws_size,
                              hipStream_t stream) {
    const float* h    = (const float*)d_in[0];
    const float* cell = (const float*)d_in[1];
    const float* x    = (const float*)d_in[2];
    const int*   ei   = (const int*)d_in[3];
    const int N_ = in_sizes[0] / D;
    const int E_ = in_sizes[3] / 2;

    WeightPtrs W;
    for (int gi = 0; gi < 4; ++gi) {             // Wg_f,bg_f,Wg_i,bg_i,...
        W.Wg[gi] = (const float*)d_in[4 + 2 * gi];
        W.bg[gi] = (const float*)d_in[5 + 2 * gi];
    }
    for (int gi = 0; gi < 4; ++gi) {             // Wl_f,bl_f,Wr_f,...
        W.Wl[gi] = (const float*)d_in[12 + 3 * gi];
        W.bl[gi] = (const float*)d_in[13 + 3 * gi];
        W.Wr[gi] = (const float*)d_in[14 + 3 * gi];
    }
    const float* gamma = (const float*)d_in[24];
    const float* beta  = (const float*)d_in[25];

    // workspace carve: agg[N*D] f32 | deg[N] f32 | WcombT[512*384] f16 | bias[512] f32
    uintptr_t base = (uintptr_t)d_ws;
    size_t off = 0;
    float* agg = (float*)(base + off); off += (size_t)N_ * D * sizeof(float);
    float* deg = (float*)(base + off);
    off += (((size_t)N_ * sizeof(float)) + 255) & ~(size_t)255;
    _Float16* WcombT = (_Float16*)(base + off); off += (size_t)CTOT * KTOT * sizeof(_Float16);
    float* biasComb  = (float*)(base + off);    off += CTOT * sizeof(float);

    // agg and deg are contiguous: zero both in one pass
    zero_ws<<<1024, 256, 0, stream>>>(agg, (long long)N_ * D + N_);
    prep_weights<<<(CTOT * KTOT + 255) / 256, 256, 0, stream>>>(W, WcombT, biasComb);
    edge_scatter<<<(E_ + 7) / 8, 256, 0, stream>>>(h, ei, agg, deg, E_, N_);
    agg_div<<<(N_ * D + 255) / 256, 256, 0, stream>>>(agg, deg, N_);
    fused_gates<<<(N_ + 31) / 32, 128, 0, stream>>>(
        h, x, agg, cell, WcombT, biasComb, gamma, beta,
        (float*)d_out, (float*)d_out + (size_t)N_ * D, N_);
}
